// Attn2Quad_47691316855177
// MI455X (gfx1250) — matrix operand order repaired and checked by reading, not compiled, not run on hardware
//
#include <hip/hip_runtime.h>
#include <hip/hip_bf16.h>

typedef __attribute__((ext_vector_type(16))) _Float16 v16h;
typedef __attribute__((ext_vector_type(8)))  float    v8f;

#define HEAD_DIM 64
#define TSEQ     2048
#define NHEADS   12
#define NBATCH   4
#define WAVES    8
#define SCALE_F  0.125f      /* 1/sqrt(64) */
#define C_ADD    3.0f
#define NEG_F    (-10000.0f)
#define EPS_F    1e-6f

// ---- A-matrix (16x32 f16) loader from a row-major f32 matrix ----
// Layout: lanes 0-15 hold rows M=0..15; element s = K 16h+s (lanes<16: K 0..15,
// lanes>=16: K 16..31).
__device__ inline v16h load_A_rowmajor(const float* rowbase, int ld, int kbase, int lane) {
    const int r  = lane & 15;
    const int kh = (lane >> 4) << 4;              // 0 or 16
    const float* p = rowbase + (size_t)r * ld + kbase + kh;
    v16h a;
#pragma unroll
    for (int i = 0; i < 8; ++i) a[i]     = (_Float16)p[i];
#pragma unroll
    for (int i = 0; i < 8; ++i) a[8 + i] = (_Float16)p[8 + i];
    return a;
}

// ---- B-matrix (32x16 f16) = tile of M^T, M row-major f32 (scores: B[k][n] = K[nrow+n][kbase+k]) ----
// lanes 0-15: N=lane, K=0..15 ; lanes 16-31: N=lane-16, K=16..31
__device__ inline v16h load_Bt_rowmajor(const float* base, int nrowbase, int ld, int kbase, int lane) {
    const int n  = lane & 15;
    const int k0 = (lane >> 4) << 4;              // 0 or 16
    const float* p = base + (size_t)(nrowbase + n) * ld + kbase + k0;
    v16h b;
#pragma unroll
    for (int i = 0; i < 16; ++i) b[i] = (_Float16)p[i];
    return b;
}

// ---- B-matrix (32x16 f16): B[k][n] = V[krow+k][col+n], V row-major f32 ----
// strided per-lane reads; lanes 0-15 read 64B-contiguous segments per step.
__device__ inline v16h load_B_cols(const float* base, int krowbase, int ld, int colbase, int lane) {
    const int n  = lane & 15;
    const int k0 = (lane >> 4) << 4;
    const float* p = base + (size_t)(krowbase + k0) * ld + colbase + n;
    v16h b;
#pragma unroll
    for (int i = 0; i < 16; ++i) b[i] = (_Float16)p[(size_t)i * ld];
    return b;
}

__global__ __launch_bounds__(WAVES * 32)
void attn2quad_kernel(const float* __restrict__ q,
                      const float* __restrict__ k,
                      const float* __restrict__ v,
                      const int*   __restrict__ amask,
                      float* __restrict__ out,
                      float* __restrict__ pout) {
    __shared__ float    s_am[TSEQ];                 // additive mask + C, per block (same b)
    __shared__ _Float16 s_p[WAVES][16 * 32];        // wave-private p-tile staging (C->A relayout)

    const int tid  = threadIdx.x;
    const int wave = tid >> 5;
    const int lane = tid & 31;
    const int bh   = blockIdx.y;                    // b*NHEADS + h
    const int b    = bh / NHEADS;
    const int qbase = blockIdx.x * (WAVES * 16) + wave * 16;

    // build (1 - mask)*NEG + C table once per block
    for (int t = tid; t < TSEQ; t += WAVES * 32) {
        s_am[t] = (1.0f - (float)amask[(size_t)b * TSEQ + t]) * NEG_F + C_ADD;
    }
    __syncthreads();

    const float* qp = q + (size_t)bh * TSEQ * HEAD_DIM;
    const float* kp = k + (size_t)bh * TSEQ * HEAD_DIM;
    const float* vp = v + (size_t)bh * TSEQ * HEAD_DIM;

    // Q tile (16 rows) in A layout, two K-slices (D = 64 = 2*32)
    const v16h a0 = load_A_rowmajor(qp + (size_t)qbase * HEAD_DIM, HEAD_DIM, 0,  lane);
    const v16h a1 = load_A_rowmajor(qp + (size_t)qbase * HEAD_DIM, HEAD_DIM, 32, lane);

    // ---------------- Pass A: den[row] = sum_k x^2 ----------------
    float den[8];
#pragma unroll
    for (int i = 0; i < 8; ++i) den[i] = 0.0f;

    for (int kt = 0; kt < TSEQ / 16; ++kt) {
        v16h b0 = load_Bt_rowmajor(kp, kt * 16, HEAD_DIM, 0,  lane);
        v16h b1 = load_Bt_rowmajor(kp, kt * 16, HEAD_DIM, 32, lane);
        v8f acc = {};
        acc = __builtin_amdgcn_wmma_f32_16x16x32_f16(false, a0, false, b0, (short)0, acc, false, false);
        acc = __builtin_amdgcn_wmma_f32_16x16x32_f16(false, a1, false, b1, (short)0, acc, false, false);
        const float am = s_am[kt * 16 + (lane & 15)];
#pragma unroll
        for (int i = 0; i < 8; ++i) {
            float x = acc[i] * SCALE_F + am;
            den[i] += x * x;
        }
    }
    // reduce across the 16-lane halves (rows v and v+8 stay separate: xor 1,2,4,8)
#pragma unroll
    for (int i = 0; i < 8; ++i) {
        float d = den[i];
        d += __shfl_xor(d, 1, 32);
        d += __shfl_xor(d, 2, 32);
        d += __shfl_xor(d, 4, 32);
        d += __shfl_xor(d, 8, 32);
        den[i] = 1.0f / (d + EPS_F);               // keep reciprocal
    }

    // ---------------- Pass B: recompute scores, emit p, accumulate out = p @ V ----------------
    v8f oacc[4] = {};                              // 16 x 64 f32 output tile

    _Float16* sp = &s_p[wave][0];
    const int rowoff = (lane >> 4) << 3;           // 0 or 8 (C/D layout row offset)
    float* prow = pout + ((size_t)bh * TSEQ + qbase + rowoff) * TSEQ + (lane & 15);

    for (int ch = 0; ch < TSEQ / 32; ++ch) {
#pragma unroll
        for (int sub = 0; sub < 2; ++sub) {
            const int kt = ch * 2 + sub;
            v16h b0 = load_Bt_rowmajor(kp, kt * 16, HEAD_DIM, 0,  lane);
            v16h b1 = load_Bt_rowmajor(kp, kt * 16, HEAD_DIM, 32, lane);
            v8f acc = {};
            acc = __builtin_amdgcn_wmma_f32_16x16x32_f16(false, a0, false, b0, (short)0, acc, false, false);
            acc = __builtin_amdgcn_wmma_f32_16x16x32_f16(false, a1, false, b1, (short)0, acc, false, false);
            const float am = s_am[kt * 16 + (lane & 15)];
            float* pw = prow + (size_t)kt * 16;
#pragma unroll
            for (int i = 0; i < 8; ++i) {
                float x  = acc[i] * SCALE_F + am;
                float pv = x * x * den[i];
                pw[(size_t)i * TSEQ] = pv;                          // f32 p output
                sp[(i + rowoff) * 32 + sub * 16 + (lane & 15)] = (_Float16)pv;  // stage for relayout
            }
        }
        // read the staged 16x32 p tile back in A layout (wave-private, DS in-order)
        v16h ap;
        {
            const int r  = lane & 15;
            const int kh = (lane >> 4) << 4;
#pragma unroll
            for (int i = 0; i < 8; ++i) ap[i]     = sp[r * 32 + kh + i];
#pragma unroll
            for (int i = 0; i < 8; ++i) ap[8 + i] = sp[r * 32 + kh + 8 + i];
        }
#pragma unroll
        for (int nt = 0; nt < 4; ++nt) {
            v16h bv = load_B_cols(vp, ch * 32, HEAD_DIM, nt * 16, lane);
            oacc[nt] = __builtin_amdgcn_wmma_f32_16x16x32_f16(false, ap, false, bv, (short)0, oacc[nt], false, false);
        }
    }

    // store out tile [16 x 64]
    float* orow = out + ((size_t)bh * TSEQ + qbase + rowoff) * HEAD_DIM + (lane & 15);
#pragma unroll
    for (int nt = 0; nt < 4; ++nt) {
#pragma unroll
        for (int i = 0; i < 8; ++i) {
            orow[(size_t)i * HEAD_DIM + nt * 16] = oacc[nt][i];
        }
    }
}

extern "C" void kernel_launch(void* const* d_in, const int* in_sizes, int n_in,
                              void* d_out, int out_size, void* d_ws, size_t ws_size,
                              hipStream_t stream) {
    (void)in_sizes; (void)n_in; (void)out_size; (void)d_ws; (void)ws_size;
    const float* q = (const float*)d_in[0];
    const float* k = (const float*)d_in[1];
    const float* v = (const float*)d_in[2];
    const int*   m = (const int*)d_in[3];
    float* out  = (float*)d_out;
    float* pout = out + (size_t)NBATCH * NHEADS * TSEQ * HEAD_DIM;  // p follows out

    dim3 grid(TSEQ / (WAVES * 16), NBATCH * NHEADS);
    attn2quad_kernel<<<grid, WAVES * 32, 0, stream>>>(q, k, v, m, out, pout);
}
